// GraphVAE_67370857005485
// MI455X (gfx1250) — matrix operand order repaired and checked by reading, not compiled, not run on hardware
//
#include <hip/hip_runtime.h>
#include <hip/hip_bf16.h>
#include <math.h>

typedef __attribute__((ext_vector_type(16))) _Float16 v16h;
typedef __attribute__((ext_vector_type(8)))  _Float16 v8h;
typedef __attribute__((ext_vector_type(8)))  float    v8f;

#define TBM 64
#define TBN 64
#define TBK 32

// ---------------------------------------------------------------------------
// CDNA5 async-copy helpers (GLOBAL_LOAD_ASYNC_TO_LDS_B128, ASYNCcnt)
// ---------------------------------------------------------------------------
__device__ __forceinline__ void async_lds_b128(unsigned lds_off, const void* gaddr) {
  asm volatile("global_load_async_to_lds_b128 %0, %1, off"
               :: "v"(lds_off), "v"(gaddr) : "memory");
}
__device__ __forceinline__ void wait_async0() {
  asm volatile("s_wait_asynccnt 0x0" ::: "memory");
}

__device__ __forceinline__ float mishf(float x) {
  float sp = (x > 20.f) ? x : log1pf(expf(x));
  return x * tanhf(sp);
}

// ---------------------------------------------------------------------------
// Tiled GEMM, all-f16 operands, f32 WMMA accumulation.
//   C[M,N] = act(A[M,K] * B[K,N] + bias) ;  B supplied TRANSPOSED: Bt[N,K]
// 128 threads (4 waves), block tile 64x64, wave tile 32x32 (2x2 WMMA 16x16x32).
// Double-buffered LDS fed by GLOBAL_LOAD_ASYNC_TO_LDS_B128.
// Requires M%64==0, N%64==0, K%32==0 (true for all GEMMs here).
// act: 0=none, 1=relu, 2=tanh.  Cf (f32) and/or Ch (f16) outputs, either null.
// ---------------------------------------------------------------------------
__global__ __launch_bounds__(128)
void gemm_f16_wmma(const _Float16* __restrict__ A, long long sA,
                   const _Float16* __restrict__ Bt, long long sB,
                   const float* __restrict__ bias,
                   float* __restrict__ Cf, _Float16* __restrict__ Ch,
                   long long sC, int M, int N, int K, int act)
{
  __shared__ alignas(128) _Float16 As[2][TBM][TBK];   // 2 x 4 KB
  __shared__ alignas(128) _Float16 Bs[2][TBN][TBK];   // 2 x 4 KB

  const int tid  = threadIdx.x;
  const int lane = tid & 31;
  const int wave = tid >> 5;
  const int wm = (wave >> 1) * 32;
  const int wn = (wave & 1) * 32;
  const int m0 = blockIdx.y * TBM;
  const int n0 = blockIdx.x * TBN;

  A  += (long long)blockIdx.z * sA;
  Bt += (long long)blockIdx.z * sB;

  // Each thread stages 2 x 16B per operand per K-step: rows rr and rr+32.
  const int rr = tid >> 2;           // 0..31
  const int cc = (tid & 3) * 8;      // 0,8,16,24 (f16 elements)
  const _Float16* gA = A  + (long long)(m0 + rr) * K + cc;
  const _Float16* gB = Bt + (long long)(n0 + rr) * K + cc;
  const long long rowK32 = (long long)32 * K;

  unsigned lA0[2], lA1[2], lB0[2], lB1[2];
  #pragma unroll
  for (int b = 0; b < 2; ++b) {
    lA0[b] = (unsigned)(size_t)&As[b][rr][cc];
    lA1[b] = (unsigned)(size_t)&As[b][rr + 32][cc];
    lB0[b] = (unsigned)(size_t)&Bs[b][rr][cc];
    lB1[b] = (unsigned)(size_t)&Bs[b][rr + 32][cc];
  }

  v8f acc[2][2] = {};
  const int fr = lane & 15;   // M index (A frag) / N index (B frag)
  const int g  = lane >> 4;   // lane-group half select

  // prologue: stage first K-tile into buffer 0
  async_lds_b128(lA0[0], gA);
  async_lds_b128(lA1[0], gA + rowK32);
  async_lds_b128(lB0[0], gB);
  async_lds_b128(lB1[0], gB + rowK32);
  wait_async0();
  __syncthreads();

  int buf = 0;
  for (int k0 = 0; k0 < K; k0 += TBK) {
    if (k0 + TBK < K) {   // stream next tile into the other buffer
      const int nb = buf ^ 1;
      async_lds_b128(lA0[nb], gA + k0 + TBK);
      async_lds_b128(lA1[nb], gA + k0 + TBK + rowK32);
      async_lds_b128(lB0[nb], gB + k0 + TBK);
      async_lds_b128(lB1[nb], gB + k0 + TBK + rowK32);
    }

    // A 16x32 f16 frag: lane group g holds K in {8g..8g+7} U {16+8g..23+8g}
    union AV { v16h v; v8h h[2]; };
    AV af[2];
    AV bf[2];
    #pragma unroll
    for (int mt = 0; mt < 2; ++mt) {
      const int row = wm + mt * 16 + fr;
      af[mt].h[0] = *(const v8h*)&As[buf][row][8 * g];
      af[mt].h[1] = *(const v8h*)&As[buf][row][16 + 8 * g];
    }
    // B 32x16 f16 frag: lane group g holds K in {8g..8g+7} U {16+8g..23+8g}
    #pragma unroll
    for (int nt = 0; nt < 2; ++nt) {
      bf[nt].h[0] = *(const v8h*)&Bs[buf][wn + nt * 16 + fr][8 * g];
      bf[nt].h[1] = *(const v8h*)&Bs[buf][wn + nt * 16 + fr][16 + 8 * g];
    }

    #pragma unroll
    for (int mt = 0; mt < 2; ++mt)
      #pragma unroll
      for (int nt = 0; nt < 2; ++nt)
        acc[mt][nt] = __builtin_amdgcn_wmma_f32_16x16x32_f16(
            false, af[mt].v, false, bf[nt].v, (short)0, acc[mt][nt], false, false);

    wait_async0();      // next tile resident (overlapped with WMMA above)
    __syncthreads();    // everyone done reading 'buf' before it is overwritten
    buf ^= 1;
  }

  // epilogue: C layout lane l -> n=l&15, vgpr r -> m=r+8*(l>>4)
  const int nf = lane & 15;
  const int mh = (lane >> 4) * 8;
  if (Cf) Cf += (long long)blockIdx.z * sC;
  if (Ch) Ch += (long long)blockIdx.z * sC;
  #pragma unroll
  for (int mt = 0; mt < 2; ++mt) {
    #pragma unroll
    for (int nt = 0; nt < 2; ++nt) {
      const int nn = n0 + wn + nt * 16 + nf;
      const float bv = bias ? bias[nn] : 0.f;
      #pragma unroll
      for (int r = 0; r < 8; ++r) {
        const int mm = m0 + wm + mt * 16 + mh + r;
        float v = acc[mt][nt][r] + bv;
        if (act == 1)      v = fmaxf(v, 0.f);
        else if (act == 2) v = tanhf(v);
        const long long idx = (long long)mm * N + nn;
        if (Cf) Cf[idx] = v;
        if (Ch) Ch[idx] = (_Float16)v;
      }
    }
  }
}

// ---------------------------------------------------------------------------
// conversion kernels (one-time per launch)
// ---------------------------------------------------------------------------
__global__ void cvt_f16_kernel(const float* __restrict__ in,
                               _Float16* __restrict__ out, int total) {
  const int i = blockIdx.x * blockDim.x + threadIdx.x;
  if (i < total) out[i] = (_Float16)in[i];
}

// src [batch][R][Cc] f32  ->  dst [batch][Cc][R] f16
__global__ void tr_f16_kernel(const float* __restrict__ src,
                              _Float16* __restrict__ dst,
                              int R, int Cc, int total) {
  const int i = blockIdx.x * blockDim.x + threadIdx.x;
  if (i >= total) return;
  const int b = i / (R * Cc);
  const int rem = i - b * (R * Cc);
  const int r = rem / Cc, c = rem - r * Cc;
  dst[((long long)b * Cc + c) * R + r] = (_Float16)src[i];
}

// d_in weight [257,256] -> transposed+padded [256,288] f16 (zero K rows >=257)
__global__ void trpad_din_kernel(const float* __restrict__ w,
                                 _Float16* __restrict__ wt) {
  const int i = blockIdx.x * blockDim.x + threadIdx.x;   // 256*288
  if (i >= 256 * 288) return;
  const int n = i / 288, k = i - n * 288;
  wt[i] = (k < 257) ? (_Float16)w[k * 256 + n] : (_Float16)0.f;
}

// ---------------------------------------------------------------------------
// BatchNorm statistics: deterministic two-stage reduction over M rows of [M,C]
// ---------------------------------------------------------------------------
__global__ void stats_partial_kernel(const float* __restrict__ a,
                                     float* __restrict__ part, int M, int C) {
  const int c = blockIdx.x * blockDim.x + threadIdx.x;
  if (c >= C) return;
  float s = 0.f, s2 = 0.f;
  for (int m = blockIdx.y; m < M; m += gridDim.y) {
    const float v = a[(long long)m * C + c];
    s += v; s2 += v * v;
  }
  part[(long long)blockIdx.y * (2 * C) + c]     = s;
  part[(long long)blockIdx.y * (2 * C) + C + c] = s2;
}

__global__ void stats_final_kernel(const float* __restrict__ part,
                                   float* __restrict__ mean,
                                   float* __restrict__ rstd,
                                   int C, int M, int P) {
  const int c = blockIdx.x * blockDim.x + threadIdx.x;
  if (c >= C) return;
  float s = 0.f, s2 = 0.f;
  for (int p = 0; p < P; ++p) {
    s  += part[(long long)p * (2 * C) + c];
    s2 += part[(long long)p * (2 * C) + C + c];
  }
  const float mu  = s / (float)M;
  const float var = s2 / (float)M - mu * mu;
  mean[c] = mu;
  rstd[c] = rsqrtf(var + 1e-5f);
}

// normalize + mish, emit any of: f16 row-major [M,C], f16 per-batch transposed
// [b,C,Nn] (for einsum B-operand), f32 row-major (for residuals).
__global__ void bn_mish_kernel(const float* __restrict__ in,
                               const float* __restrict__ mean,
                               const float* __restrict__ rstd,
                               _Float16* __restrict__ out_rm,
                               _Float16* __restrict__ out_tr,
                               float* __restrict__ out_f32,
                               int total, int C, int Nn) {
  const int i = blockIdx.x * blockDim.x + threadIdx.x;
  if (i >= total) return;
  const int c = i % C;
  const int m = i / C;
  const float y = mishf((in[i] - mean[c]) * rstd[c]);
  if (out_rm)  out_rm[i] = (_Float16)y;
  if (out_tr) {
    const int b = m / Nn, n = m - b * Nn;
    out_tr[((long long)b * C + c) * Nn + n] = (_Float16)y;
  }
  if (out_f32) out_f32[i] = y;
}

__global__ void add_kernel(const float* __restrict__ a,
                           const float* __restrict__ b,
                           float* __restrict__ out, int total) {
  const int i = blockIdx.x * blockDim.x + threadIdx.x;
  if (i < total) out[i] = a[i] + b[i];
}

// hbar[b,e] = mean_n((r[b,n,e]-mean[e])*rstd[e]) -> f16   (B=64,N=256,E=256)
__global__ void hbar_kernel(const float* __restrict__ r,
                            const float* __restrict__ mean,
                            const float* __restrict__ rstd,
                            _Float16* __restrict__ hbar) {
  const int e = threadIdx.x;
  const int b = blockIdx.x;
  float s = 0.f;
  for (int n = 0; n < 256; ++n)
    s += r[((long long)(b * 256 + n)) * 256 + e];
  hbar[b * 256 + e] = (_Float16)((s * (1.f / 256.f) - mean[e]) * rstd[e]);
}

// alpha_cumprod[t[b]] via serial cumprod of (1 - linspace(1e-4, 0.02, 1000))
__global__ void ac_kernel(const int* __restrict__ t, float* __restrict__ ac) {
  const int b = threadIdx.x;
  if (b >= 64) return;
  const int tb = t[b];
  float prod = 1.f;
  for (int j = 0; j <= tb; ++j) {
    const float beta = 1e-4f + (0.02f - 1e-4f) * ((float)j / 999.f);
    prod *= (1.f - beta);
  }
  ac[b] = prod;
}

// clip logstd, z = mu + eps*exp(ls); z_t -> padded f16 [64,288]; z -> f16
__global__ void latent_kernel(const float* __restrict__ mu,
                              const float* __restrict__ lsraw,
                              const float* __restrict__ eps_z,
                              const float* __restrict__ eps_n,
                              const float* __restrict__ ac,
                              const int* __restrict__ t,
                              float* __restrict__ logstd_out,
                              float* __restrict__ z_out,
                              _Float16* __restrict__ z16,
                              _Float16* __restrict__ val16) {
  const int i = blockIdx.x * blockDim.x + threadIdx.x;
  if (i >= 64 * 256) return;
  const int b = i >> 8, e = i & 255;
  const float ls = fminf(2.f, fmaxf(-10.f, lsraw[i]));
  logstd_out[i] = ls;
  const float z = mu[i] + eps_z[i] * expf(ls);
  z_out[i] = z;
  z16[i]   = (_Float16)z;     // z_recon == z algebraically
  const float a  = ac[b];
  const float zt = sqrtf(a) * z + sqrtf(1.f - a) * eps_n[i];
  val16[(long long)b * 288 + e] = (_Float16)zt;
  if (e == 0) {
    val16[(long long)b * 288 + 256] = (_Float16)(float)t[b];
    for (int j = 257; j < 288; ++j) val16[(long long)b * 288 + j] = (_Float16)0.f;
  }
}

// ---------------------------------------------------------------------------
// host orchestration
// ---------------------------------------------------------------------------
static void launch_gemm(hipStream_t s, const _Float16* A, long long sA,
                        const _Float16* Bt, long long sB, const float* bias,
                        float* Cf, _Float16* Ch, long long sC,
                        int M, int N, int K, int batch, int act) {
  dim3 grid(N / TBN, M / TBM, batch), block(128);
  gemm_f16_wmma<<<grid, block, 0, s>>>(A, sA, Bt, sB, bias, Cf, Ch, sC, M, N, K, act);
}

static void launch_stats(hipStream_t s, const float* a, float* part,
                         float* mean, float* rstd, int M, int C) {
  dim3 g1((C + 255) / 256, 32);
  stats_partial_kernel<<<g1, 256, 0, s>>>(a, part, M, C);
  stats_final_kernel<<<(C + 255) / 256, 256, 0, s>>>(part, mean, rstd, C, M, 32);
}

extern "C" void kernel_launch(void* const* d_in, const int* in_sizes, int n_in,
                              void* d_out, int out_size, void* d_ws, size_t ws_size,
                              hipStream_t stream) {
  (void)in_sizes; (void)n_in; (void)out_size; (void)ws_size;
  const int Bsz = 64, Nn = 256, D = 128, H = 512, E = 256;
  const int M = Bsz * Nn;   // 16384

  const float* x     = (const float*)d_in[0];
  const float* adj   = (const float*)d_in[1];
  const int*   t     = (const int*)  d_in[2];
  const float* eps_z = (const float*)d_in[3];
  const float* eps_n = (const float*)d_in[4];
  auto W  = [&](int i) { return (const float*)d_in[5 + 2 * i]; };
  auto Bb = [&](int i) { return (const float*)d_in[5 + 2 * i + 1]; };
  // 0 b1.w1, 1 b1.w2, 2 b1.sc, 3 b2.w1, 4 b2.w2, 5 b3.w1, 6 b3.w2, 7 b3.sc,
  // 8 mu, 9 logstd, 10 d_in, 11 d_m0, 12 d_m1, 13 d_out, 14 dec1..17 dec4
  static const int WK[18] = {128,512,128,512,512,512,256,512,256,256,257,256,256,256,256,512,512,512};
  static const int WN[18] = {512,512,512,512,512,256,256,256,256,256,256,256,256,256,512,512,512,32768};

  // ---- output layout (recon_x, mu, logstd, predicted_z, z) ----
  float* out   = (float*)d_out;
  float* recon = out;
  float* o_mu  = out + 2097152;
  float* o_ls  = o_mu + 16384;
  float* o_pz  = o_ls + 16384;
  float* o_z   = o_pz + 16384;

  // ---- workspace layout ----
  float* ws = (float*)d_ws;
  const long long BIG = (long long)M * H;   // 8388608
  float* P     = ws;                         // f32 preact / residual
  float* G     = P + BIG;                    // f32 scratch
  float* part  = G + BIG;                    // 32*2*512
  float* meanb = part + 32 * 1024;
  float* rstdb = meanb + 512;
  float* lsraw = rstdb + 512;                // 64*256
  float* acb   = lsraw + 16384;              // 64
  _Float16* hp = (_Float16*)(acb + 64);
  _Float16* adj16 = hp;                hp += 4194304;          // [B,256,256]
  _Float16* x16   = hp;                hp += 2097152;          // [M,128] rm
  _Float16* x16t  = hp;                hp += 2097152;          // [B,128,256]
  _Float16* T16   = hp;                hp += BIG;              // einsum out (rm)
  _Float16* U16   = hp;                hp += BIG;              // act (per-batch transposed)
  _Float16* R16   = hp;                hp += BIG;              // act (rm, for shortcut A)
  _Float16* hbar16= hp;                hp += 16384;
  _Float16* z16   = hp;                hp += 16384;
  _Float16* val16 = hp;                hp += 18432;            // [64,288]
  _Float16* mlpA  = hp;                hp += 16384;
  _Float16* mlpB  = hp;                hp += 16384;
  _Float16* dec16 = hp;                hp += 32768;            // [64,512]
  _Float16* wT[18];
  for (int i = 0; i < 18; ++i) {
    wT[i] = hp;
    hp += (long long)((i == 10) ? 288 : WK[i]) * WN[i];
  }

  // ---- one-time converts / transposes (f32 -> f16) ----
  cvt_f16_kernel<<<4194304 / 256, 256, 0, stream>>>(adj, adj16, 4194304);
  cvt_f16_kernel<<<2097152 / 256, 256, 0, stream>>>(x, x16, 2097152);
  tr_f16_kernel<<<2097152 / 256, 256, 0, stream>>>(x, x16t, 256, 128, 2097152);
  for (int i = 0; i < 18; ++i) {
    if (i == 10) {
      trpad_din_kernel<<<(256 * 288) / 256, 256, 0, stream>>>(W(10), wT[10]);
    } else {
      const int tot = WK[i] * WN[i];
      tr_f16_kernel<<<(tot + 255) / 256, 256, 0, stream>>>(W(i), wT[i], WK[i], WN[i], tot);
    }
  }

  const long long sAdj = (long long)Nn * Nn;

  // ================= encoder block 1 (D -> H, linear shortcut) ==============
  launch_gemm(stream, adj16, sAdj, x16t, (long long)D * Nn, nullptr,
              nullptr, T16, (long long)Nn * D, Nn, D, Nn, Bsz, 0);       // t0
  launch_gemm(stream, T16, 0, wT[0], 0, Bb(0), P, nullptr, 0, M, H, D, 1, 0);
  launch_stats(stream, P, part, meanb, rstdb, M, H);
  bn_mish_kernel<<<(M * H) / 256, 256, 0, stream>>>(P, meanb, rstdb,
      nullptr, U16, nullptr, M * H, H, Nn);                              // a1 (tr)
  launch_gemm(stream, adj16, sAdj, U16, (long long)H * Nn, nullptr,
              nullptr, T16, (long long)Nn * H, Nn, H, Nn, Bsz, 0);       // t1
  launch_gemm(stream, T16, 0, wT[1], 0, Bb(1), P, nullptr, 0, M, H, H, 1, 0);
  launch_stats(stream, P, part, meanb, rstdb, M, H);
  bn_mish_kernel<<<(M * H) / 256, 256, 0, stream>>>(P, meanb, rstdb,
      nullptr, nullptr, G, M * H, H, Nn);                                // a2 (f32)
  launch_gemm(stream, x16, 0, wT[2], 0, Bb(2), P, nullptr, 0, M, H, D, 1, 0); // sc
  add_kernel<<<(M * H) / 256, 256, 0, stream>>>(G, P, G, M * H);
  launch_stats(stream, G, part, meanb, rstdb, M, H);
  bn_mish_kernel<<<(M * H) / 256, 256, 0, stream>>>(G, meanb, rstdb,
      nullptr, U16, P, M * H, H, Nn);        // block1 out: tr + f32 (residual)

  // ================= encoder block 2 (H -> H, identity shortcut) ============
  launch_gemm(stream, adj16, sAdj, U16, (long long)H * Nn, nullptr,
              nullptr, T16, (long long)Nn * H, Nn, H, Nn, Bsz, 0);
  launch_gemm(stream, T16, 0, wT[3], 0, Bb(3), G, nullptr, 0, M, H, H, 1, 0);
  launch_stats(stream, G, part, meanb, rstdb, M, H);
  bn_mish_kernel<<<(M * H) / 256, 256, 0, stream>>>(G, meanb, rstdb,
      nullptr, U16, nullptr, M * H, H, Nn);
  launch_gemm(stream, adj16, sAdj, U16, (long long)H * Nn, nullptr,
              nullptr, T16, (long long)Nn * H, Nn, H, Nn, Bsz, 0);
  launch_gemm(stream, T16, 0, wT[4], 0, Bb(4), G, nullptr, 0, M, H, H, 1, 0);
  launch_stats(stream, G, part, meanb, rstdb, M, H);
  bn_mish_kernel<<<(M * H) / 256, 256, 0, stream>>>(G, meanb, rstdb,
      nullptr, nullptr, G, M * H, H, Nn);                 // a2 in-place
  add_kernel<<<(M * H) / 256, 256, 0, stream>>>(G, P, G, M * H);
  launch_stats(stream, G, part, meanb, rstdb, M, H);
  bn_mish_kernel<<<(M * H) / 256, 256, 0, stream>>>(G, meanb, rstdb,
      R16, U16, nullptr, M * H, H, Nn);      // block2 out: rm (for sc) + tr

  // ================= encoder block 3 (H -> E, linear shortcut) ==============
  launch_gemm(stream, adj16, sAdj, U16, (long long)H * Nn, nullptr,
              nullptr, T16, (long long)Nn * H, Nn, H, Nn, Bsz, 0);
  launch_gemm(stream, T16, 0, wT[5], 0, Bb(5), P, nullptr, 0, M, E, H, 1, 0);
  launch_stats(stream, P, part, meanb, rstdb, M, E);
  bn_mish_kernel<<<(M * E) / 256, 256, 0, stream>>>(P, meanb, rstdb,
      nullptr, U16, nullptr, M * E, E, Nn);
  launch_gemm(stream, adj16, sAdj, U16, (long long)E * Nn, nullptr,
              nullptr, T16, (long long)Nn * E, Nn, E, Nn, Bsz, 0);
  launch_gemm(stream, T16, 0, wT[6], 0, Bb(6), P, nullptr, 0, M, E, E, 1, 0);
  launch_stats(stream, P, part, meanb, rstdb, M, E);
  bn_mish_kernel<<<(M * E) / 256, 256, 0, stream>>>(P, meanb, rstdb,
      nullptr, nullptr, P, M * E, E, Nn);                 // a2 in-place
  launch_gemm(stream, R16, 0, wT[7], 0, Bb(7), G, nullptr, 0, M, E, H, 1, 0); // sc
  add_kernel<<<(M * E) / 256, 256, 0, stream>>>(P, G, P, M * E);
  launch_stats(stream, P, part, meanb, rstdb, M, E);      // final bn (no mish)
  hbar_kernel<<<Bsz, E, 0, stream>>>(P, meanb, rstdb, hbar16);  // + mean over N

  // ================= latent head =================
  launch_gemm(stream, hbar16, 0, wT[8], 0, Bb(8), o_mu, nullptr, 0, Bsz, E, E, 1, 0);
  launch_gemm(stream, hbar16, 0, wT[9], 0, Bb(9), lsraw, nullptr, 0, Bsz, E, E, 1, 0);
  ac_kernel<<<1, 64, 0, stream>>>(t, acb);
  latent_kernel<<<(Bsz * E) / 256, 256, 0, stream>>>(
      o_mu, lsraw, eps_z, eps_n, acb, t, o_ls, o_z, z16, val16);

  // ================= diffusion noise-predictor MLP =================
  launch_gemm(stream, val16, 0, wT[10], 0, Bb(10), nullptr, mlpA, 0, Bsz, 256, 288, 1, 0);
  launch_gemm(stream, mlpA, 0, wT[11], 0, Bb(11), nullptr, mlpB, 0, Bsz, 256, 256, 1, 1);
  launch_gemm(stream, mlpB, 0, wT[12], 0, Bb(12), nullptr, mlpA, 0, Bsz, 256, 256, 1, 1);
  launch_gemm(stream, mlpA, 0, wT[13], 0, Bb(13), o_pz, nullptr, 0, Bsz, E, 256, 1, 0);

  // ================= decoder (z_recon == z) =================
  launch_gemm(stream, z16, 0, wT[14], 0, Bb(14), P, nullptr, 0, Bsz, H, E, 1, 0);
  launch_stats(stream, P, part, meanb, rstdb, Bsz, H);
  bn_mish_kernel<<<(Bsz * H) / 256, 256, 0, stream>>>(P, meanb, rstdb,
      dec16, nullptr, nullptr, Bsz * H, H, 1);
  launch_gemm(stream, dec16, 0, wT[15], 0, Bb(15), P, nullptr, 0, Bsz, H, H, 1, 0);
  launch_stats(stream, P, part, meanb, rstdb, Bsz, H);
  bn_mish_kernel<<<(Bsz * H) / 256, 256, 0, stream>>>(P, meanb, rstdb,
      dec16, nullptr, nullptr, Bsz * H, H, 1);
  launch_gemm(stream, dec16, 0, wT[16], 0, Bb(16), P, nullptr, 0, Bsz, H, H, 1, 0);
  launch_stats(stream, P, part, meanb, rstdb, Bsz, H);
  bn_mish_kernel<<<(Bsz * H) / 256, 256, 0, stream>>>(P, meanb, rstdb,
      dec16, nullptr, nullptr, Bsz * H, H, 1);
  // dec4: [64,512] @ [512,32768] + bias, tanh -> recon_x
  launch_gemm(stream, dec16, 0, wT[17], 0, Bb(17), recon, nullptr, 0,
              Bsz, Nn * D, H, 1, 2);
}
